// LinearAttention_19275813224828
// MI455X (gfx1250) — matrix operand order repaired and checked by reading, not compiled, not run on hardware
//
#include <hip/hip_runtime.h>
#include <hip/hip_bf16.h>
#include <math.h>

typedef __attribute__((ext_vector_type(16))) __bf16 v16bf;
typedef __attribute__((ext_vector_type(8)))  float  v8f;

namespace {
constexpr int   kB      = 4;
constexpr int   kC      = 256;
constexpr int   kHeads  = 8;
constexpr int   kKD     = 64;
constexpr int   kVD     = 64;
constexpr int   kN      = 128 * 128;      // 16384 tokens
constexpr int   kM      = kHeads * kKD;   // 512
constexpr int   kBH     = kB * kHeads;    // 32
constexpr int   kRows   = kBH * kKD;      // 2048 k-rows for N-softmax
constexpr int   kSplit  = 8;              // K-split for ctx GEMM
constexpr float kEps    = 1e-5f;
constexpr float kQScale = 0.35355339059327373f; // 64^-0.25
} // namespace

// ---------------------------------------------------------------------------
// CDNA5 async copy helpers: per-lane 16B global -> LDS, tracked by ASYNCcnt.
// ---------------------------------------------------------------------------
__device__ __forceinline__ void async_copy_b128(const void* gsrc, void* lds_dst) {
  // Low 32 bits of the generic (flat) address of a __shared__ object are the
  // LDS byte offset (aperture base has zero low bits).
  const unsigned ldst = (unsigned)(size_t)lds_dst;
  asm volatile("global_load_async_to_lds_b128 %0, %1, off"
               :: "v"(ldst), "v"(gsrc)
               : "memory");
}
__device__ __forceinline__ void wait_async0() {
  asm volatile("s_wait_asynccnt 0" ::: "memory");
}

// ---------------------------------------------------------------------------
// WMMA helpers (CDNA5 wave32, 16x16x32 bf16, f32 accumulate)
// LDS tiles use a row stride of 40 bf16 (80B) -> fragment b128 reads are
// bank-conflict free across the 16-lane halves.
// ---------------------------------------------------------------------------
__device__ __forceinline__ v8f wmma_bf16(const v16bf& a, const v16bf& b, const v8f& c) {
  return __builtin_amdgcn_wmma_f32_16x16x32_bf16(false, a, false, b, (short)0, c,
                                                 false, false);
}

// A-matrix 16x32 bf16 fragment: lane m holds row (mBase + lane&15);
// lanes 0-15 get K=0..7 / 16..23, lanes 16-31 get K=8..15 / 24..31.
__device__ __forceinline__ v16bf frag_a(const __bf16* ls, int mBase) {
  const int lane = threadIdx.x & 31;
  const __bf16* p = ls + (mBase + (lane & 15)) * 40 + ((lane >> 4) << 3);
  v16bf a;
#pragma unroll
  for (int j = 0; j < 8; ++j) a[j] = p[j];
#pragma unroll
  for (int j = 0; j < 8; ++j) a[8 + j] = p[16 + j];
  return a;
}

// B-matrix 32x16 bf16 fragment from an LDS tile stored transposed
// (ls[col*40 + k]): lane holds col (nBase + lane&15), K = 8*(lane>>4)+0..7 / 16+8*(lane>>4)+0..7.
__device__ __forceinline__ v16bf frag_b(const __bf16* ls, int nBase) {
  const int lane = threadIdx.x & 31;
  const __bf16* p = ls + (nBase + (lane & 15)) * 40 + ((lane >> 4) << 3);
  v16bf b;
#pragma unroll
  for (int j = 0; j < 8; ++j) b[j] = p[j];
#pragma unroll
  for (int j = 0; j < 8; ++j) b[8 + j] = p[16 + j];
  return b;
}

// ---------------------------------------------------------------------------
// K1: QKV projection  out[b, m, n] = bf16( W[m, :] . x[b, :, n] + bias[m] )
// 128x128 block tile, K-step 32, 8 waves (4x2), wave tile 32x64 (2x4 frags).
// ---------------------------------------------------------------------------
__global__ __launch_bounds__(256)
void qkv_proj_kernel(const float* __restrict__ X, const float* __restrict__ W,
                     const float* __restrict__ bias, __bf16* __restrict__ out) {
  __shared__ alignas(16) __bf16 lsA[128 * 40];
  __shared__ alignas(16) __bf16 lsB[128 * 40];
  const int n0 = blockIdx.x * 128;
  const int m0 = blockIdx.y * 128;
  const int b  = blockIdx.z;
  const float* xb = X + (size_t)b * kC * kN;
  const int t = threadIdx.x;
  const int lane = t & 31, wid = t >> 5;
  const int wm = (wid >> 1) * 32, wn = (wid & 1) * 64;
  const int arow = t >> 1, acb = (t & 1) * 16;   // A tile 128x32, 16 elems/thread
  const int brow = t >> 3, bnb = (t & 7) * 16;   // B tile 32x128, 16 elems/thread
  v8f acc[2][4] = {};
  for (int k0 = 0; k0 < kC; k0 += 32) {
    __syncthreads();
    const float* ga = W + (size_t)(m0 + arow) * kC + k0 + acb;
#pragma unroll
    for (int i = 0; i < 16; ++i) lsA[arow * 40 + acb + i] = (__bf16)ga[i];
    const float* gb = xb + (size_t)(k0 + brow) * kN + n0 + bnb;
#pragma unroll
    for (int i = 0; i < 16; ++i) lsB[(bnb + i) * 40 + brow] = (__bf16)gb[i];
    if (k0 + 32 < kC) {             // global_prefetch_b8 for next K-step
      __builtin_prefetch(ga + 32, 0, 0);
      __builtin_prefetch(gb + (size_t)32 * kN, 0, 0);
    }
    __syncthreads();
    v16bf af[2], bfr[4];
#pragma unroll
    for (int i = 0; i < 2; ++i) af[i] = frag_a(lsA, wm + i * 16);
#pragma unroll
    for (int j = 0; j < 4; ++j) bfr[j] = frag_b(lsB, wn + j * 16);
#pragma unroll
    for (int i = 0; i < 2; ++i)
#pragma unroll
      for (int j = 0; j < 4; ++j)
        acc[i][j] = wmma_bf16(af[i], bfr[j], acc[i][j]);
  }
  const int hf = lane >> 4, nl = lane & 15;
#pragma unroll
  for (int i = 0; i < 2; ++i)
#pragma unroll
    for (int j = 0; j < 4; ++j)
#pragma unroll
      for (int r = 0; r < 8; ++r) {
        const int row = m0 + wm + i * 16 + hf * 8 + r;
        const int col = n0 + wn + j * 16 + nl;
        out[((size_t)b * kM + row) * kN + col] = (__bf16)(acc[i][j][r] + bias[row]);
      }
}

// ---------------------------------------------------------------------------
// K2: LayerNorm over the per-head dim (64) of [BH*64, N]; one thread per
// column (coalesced across adjacent n). mode 1: + q-scale + softmax over dim.
// ---------------------------------------------------------------------------
__global__ __launch_bounds__(256)
void ln_kernel(__bf16* __restrict__ buf, const float* __restrict__ g,
               const float* __restrict__ be, int softmax_dim) {
  const int id = blockIdx.x * blockDim.x + threadIdx.x;
  const int n  = id & (kN - 1);
  const int bh = id >> 14;                  // id / kN
  if (bh >= kBH) return;
  __bf16* p = buf + ((size_t)bh * kKD) * kN + n;
  float s = 0.f, ss = 0.f;
  for (int d = 0; d < kKD; ++d) {
    const float v = (float)p[(size_t)d * kN];
    s += v; ss += v * v;
  }
  const float mu   = s * (1.f / kKD);
  const float var  = ss * (1.f / kKD) - mu * mu;
  const float rstd = rsqrtf(var + kEps);
  if (!softmax_dim) {
    for (int d = 0; d < kKD; ++d) {
      const float v = (float)p[(size_t)d * kN];
      p[(size_t)d * kN] = (__bf16)(((v - mu) * rstd) * g[d] + be[d]);
    }
    return;
  }
  float mx = -1e30f;
  for (int d = 0; d < kKD; ++d) {
    const float v = (float)p[(size_t)d * kN];
    const float y = (((v - mu) * rstd) * g[d] + be[d]) * kQScale;
    mx = fmaxf(mx, y);
  }
  float sum = 0.f;
  for (int d = 0; d < kKD; ++d) {
    const float v = (float)p[(size_t)d * kN];
    const float y = (((v - mu) * rstd) * g[d] + be[d]) * kQScale;
    sum += __expf(y - mx);
  }
  const float inv = 1.f / sum;
  for (int d = 0; d < kKD; ++d) {
    const float v = (float)p[(size_t)d * kN];
    const float y = (((v - mu) * rstd) * g[d] + be[d]) * kQScale;
    p[(size_t)d * kN] = (__bf16)(__expf(y - mx) * inv);
  }
}

// ---------------------------------------------------------------------------
// K3: per-row max and 1/sum(exp(x-max)) over N for the k N-softmax.
// ---------------------------------------------------------------------------
__global__ __launch_bounds__(256)
void k_rowstats_kernel(const __bf16* __restrict__ kb, float* __restrict__ rmax,
                       float* __restrict__ rinv) {
  __shared__ float red[256];
  const int row = blockIdx.x;
  const __bf16* p = kb + (size_t)row * kN;
  const int t = threadIdx.x;
  float mx = -1e30f;
  for (int i = t; i < kN; i += 256) mx = fmaxf(mx, (float)p[i]);
  red[t] = mx; __syncthreads();
  for (int s = 128; s > 0; s >>= 1) {
    if (t < s) red[t] = fmaxf(red[t], red[t + s]);
    __syncthreads();
  }
  mx = red[0]; __syncthreads();
  float sum = 0.f;
  for (int i = t; i < kN; i += 256) sum += __expf((float)p[i] - mx);
  red[t] = sum; __syncthreads();
  for (int s = 128; s > 0; s >>= 1) {
    if (t < s) red[t] += red[t + s];
    __syncthreads();
  }
  if (t == 0) { rmax[row] = mx; rinv[row] = 1.f / red[0]; }
}

// ---------------------------------------------------------------------------
// K4: ctx_part[split,bh] = softmaxN(k)[64,Kchunk] x v^T[Kchunk,64].
// Double-buffered LDS, one barrier per K-step. v tile goes global->LDS via
// GLOBAL_LOAD_ASYNC_TO_LDS_B128 (ASYNCcnt); k tile gets exp((k-max))*inv
// applied on the VALU while staging. 8 waves (4x2), wave tile 16x32.
// ---------------------------------------------------------------------------
__global__ __launch_bounds__(256)
void ctx_gemm_kernel(const __bf16* __restrict__ kb, const __bf16* __restrict__ vb,
                     const float* __restrict__ rmax, const float* __restrict__ rinv,
                     float* __restrict__ ctx_part) {
  __shared__ alignas(16) __bf16 lsA[2][64 * 40];
  __shared__ alignas(16) __bf16 lsB[2][64 * 40];
  const int split = blockIdx.x;   // 0..kSplit-1
  const int bh    = blockIdx.y;   // 0..kBH-1
  const int t = threadIdx.x;
  const int lane = t & 31, wid = t >> 5;
  const int wm = (wid >> 1) * 16, wn = (wid & 1) * 32;
  const int arow = t >> 2, acb = (t & 3) * 8;   // 64x32 tile, 8 elems (16B)/thread
  const float amax = rmax[bh * 64 + arow];
  const float ainv = rinv[bh * 64 + arow];
  const __bf16* ka = kb + ((size_t)bh * 64 + arow) * kN;   // k row kd = arow
  const __bf16* va = vb + ((size_t)bh * 64 + arow) * kN;   // v row vd = arow
  v8f acc[2] = {};
  const int nbeg  = split * (kN / kSplit);
  const int steps = (kN / kSplit) / 32;

  auto stage = [&](int buf, int n0) {
    // B tile (v): pure 16B copy -> async direct to LDS, lsB[vd*40 + k].
    async_copy_b128(va + n0 + acb, &lsB[buf][arow * 40 + acb]);
    // A tile (k): apply N-softmax exp while staging.
#pragma unroll
    for (int i = 0; i < 8; ++i) {
      const float v = (float)ka[n0 + acb + i];
      lsA[buf][arow * 40 + acb + i] = (__bf16)(__expf(v - amax) * ainv);
    }
    __builtin_prefetch(ka + n0 + 32 + acb, 0, 0);
  };

  stage(0, nbeg);
  for (int it = 0; it < steps; ++it) {
    const int buf = it & 1;
    wait_async0();          // my async b128 into lsB[buf] has landed
    __syncthreads();        // everyone's stores/asyncs for buf are visible
    if (it + 1 < steps) stage(buf ^ 1, nbeg + (it + 1) * 32);  // overlap
    const v16bf af = frag_a(lsA[buf], wm);
#pragma unroll
    for (int j = 0; j < 2; ++j) {
      const v16bf bfr = frag_b(lsB[buf], wn + j * 16);
      acc[j] = wmma_bf16(af, bfr, acc[j]);
    }
  }
  const int hf = lane >> 4, nl = lane & 15;
  float* outp = ctx_part + ((size_t)split * kBH + bh) * (kKD * kVD);
#pragma unroll
  for (int j = 0; j < 2; ++j)
#pragma unroll
    for (int r = 0; r < 8; ++r) {
      const int kd = wm + hf * 8 + r;
      const int vd = wn + j * 16 + nl;
      outp[kd * 64 + vd] = acc[j][r];
    }
}

// ---------------------------------------------------------------------------
// K5: reduce K-splits and emit ctx^T (vd-major) in bf16 for the next GEMM.
// ---------------------------------------------------------------------------
__global__ __launch_bounds__(256)
void ctx_reduce_kernel(const float* __restrict__ ctx_part, __bf16* __restrict__ ctxT) {
  const int bh = blockIdx.x;
  for (int e = threadIdx.x; e < kKD * kVD; e += 256) {
    const int kd = e >> 6, vd = e & 63;
    float s = 0.f;
#pragma unroll
    for (int sp = 0; sp < kSplit; ++sp)
      s += ctx_part[((size_t)sp * kBH + bh) * (kKD * kVD) + e];
    ctxT[(size_t)bh * (kKD * kVD) + vd * 64 + kd] = (__bf16)s;
  }
}

// ---------------------------------------------------------------------------
// K6: attn[bh, vd, n] = ctx^T[vd, kd] x q[kd, n].  M=64, K=64, N tile 128.
// ctx^T tile is a pure 16B copy -> async to LDS. 8 waves (4x2), wave 16x64.
// ---------------------------------------------------------------------------
__global__ __launch_bounds__(256)
void attn_out_kernel(const __bf16* __restrict__ ctxT, const __bf16* __restrict__ qb,
                     __bf16* __restrict__ attn) {
  __shared__ alignas(16) __bf16 lsA[64 * 40];
  __shared__ alignas(16) __bf16 lsB[128 * 40];
  const int n0 = blockIdx.x * 128;
  const int bh = blockIdx.y;
  const int t = threadIdx.x;
  const int lane = t & 31, wid = t >> 5;
  const int wm = (wid >> 1) * 16, wn = (wid & 1) * 64;
  const __bf16* A  = ctxT + (size_t)bh * (kKD * kVD);   // [vd][kd]
  const __bf16* Bq = qb + (size_t)bh * 64 * kN;         // [kd][n]
  const int arow = t >> 2, acb = (t & 3) * 8;   // A tile 64x32
  const int brow = t >> 3, bnb = (t & 7) * 16;  // B tile 32x128
  v8f acc[4] = {};
  for (int k0 = 0; k0 < kKD; k0 += 32) {
    __syncthreads();
    async_copy_b128(A + arow * 64 + k0 + acb, &lsA[arow * 40 + acb]);
    const __bf16* gb = Bq + (size_t)(k0 + brow) * kN + n0 + bnb;
#pragma unroll
    for (int i = 0; i < 16; ++i)
      lsB[(bnb + i) * 40 + brow] = gb[i];
    wait_async0();
    __syncthreads();
    const v16bf af = frag_a(lsA, wm);
#pragma unroll
    for (int j = 0; j < 4; ++j) {
      const v16bf bfr = frag_b(lsB, wn + j * 16);
      acc[j] = wmma_bf16(af, bfr, acc[j]);
    }
  }
  const int hf = lane >> 4, nl = lane & 15;
#pragma unroll
  for (int j = 0; j < 4; ++j)
#pragma unroll
    for (int r = 0; r < 8; ++r) {
      const int vd  = wm + hf * 8 + r;
      const int col = n0 + wn + j * 16 + nl;
      attn[((size_t)bh * 64 + vd) * kN + col] = (__bf16)acc[j][r];
    }
}

// ---------------------------------------------------------------------------
// K7: output projection  Y[b, c, n] = Wo[c, :] . attn[b, :, n] + bo[c] (f32).
// ---------------------------------------------------------------------------
__global__ __launch_bounds__(256)
void out_proj_kernel(const __bf16* __restrict__ attn, const float* __restrict__ Wo,
                     const float* __restrict__ bo, float* __restrict__ Y) {
  __shared__ alignas(16) __bf16 lsA[128 * 40];
  __shared__ alignas(16) __bf16 lsB[128 * 40];
  const int n0 = blockIdx.x * 128;
  const int m0 = blockIdx.y * 128;
  const int b  = blockIdx.z;
  const __bf16* Ba = attn + (size_t)b * kM * kN;
  const int t = threadIdx.x;
  const int lane = t & 31, wid = t >> 5;
  const int wm = (wid >> 1) * 32, wn = (wid & 1) * 64;
  const int arow = t >> 1, acb = (t & 1) * 16;
  const int brow = t >> 3, bnb = (t & 7) * 16;
  v8f acc[2][4] = {};
  for (int k0 = 0; k0 < kM; k0 += 32) {
    __syncthreads();
    const float* ga = Wo + (size_t)(m0 + arow) * kM + k0 + acb;
#pragma unroll
    for (int i = 0; i < 16; ++i) lsA[arow * 40 + acb + i] = (__bf16)ga[i];
    const __bf16* gb = Ba + (size_t)(k0 + brow) * kN + n0 + bnb;
#pragma unroll
    for (int i = 0; i < 16; ++i) lsB[(bnb + i) * 40 + brow] = gb[i];
    if (k0 + 32 < kM) {
      __builtin_prefetch(ga + 32, 0, 0);
      __builtin_prefetch(gb + (size_t)32 * kN, 0, 0);
    }
    __syncthreads();
    v16bf af[2], bfr[4];
#pragma unroll
    for (int i = 0; i < 2; ++i) af[i] = frag_a(lsA, wm + i * 16);
#pragma unroll
    for (int j = 0; j < 4; ++j) bfr[j] = frag_b(lsB, wn + j * 16);
#pragma unroll
    for (int i = 0; i < 2; ++i)
#pragma unroll
      for (int j = 0; j < 4; ++j)
        acc[i][j] = wmma_bf16(af[i], bfr[j], acc[i][j]);
  }
  const int hf = lane >> 4, nl = lane & 15;
#pragma unroll
  for (int i = 0; i < 2; ++i)
#pragma unroll
    for (int j = 0; j < 4; ++j)
#pragma unroll
      for (int r = 0; r < 8; ++r) {
        const int row = m0 + wm + i * 16 + hf * 8 + r;
        const int col = n0 + wn + j * 16 + nl;
        Y[((size_t)b * kC + row) * kN + col] = acc[i][j][r] + bo[row];
      }
}

// ---------------------------------------------------------------------------
extern "C" void kernel_launch(void* const* d_in, const int* in_sizes, int n_in,
                              void* d_out, int out_size, void* d_ws, size_t ws_size,
                              hipStream_t stream) {
  (void)in_sizes; (void)n_in; (void)out_size; (void)ws_size;
  const float* x    = (const float*)d_in[0];
  const float* wq   = (const float*)d_in[1];
  const float* bq   = (const float*)d_in[2];
  const float* wk   = (const float*)d_in[3];
  const float* bk   = (const float*)d_in[4];
  const float* wv   = (const float*)d_in[5];
  const float* bv   = (const float*)d_in[6];
  const float* wo   = (const float*)d_in[7];
  const float* bo   = (const float*)d_in[8];
  const float* g_q  = (const float*)d_in[9];
  const float* be_q = (const float*)d_in[10];
  const float* g_k  = (const float*)d_in[11];
  const float* be_k = (const float*)d_in[12];
  const float* g_v  = (const float*)d_in[13];
  const float* be_v = (const float*)d_in[14];

  // Workspace layout (bf16 intermediates): qb/kb/vb/attn 64MB each,
  // ctx partials 4MB, ctxT 256KB, row stats 16KB  (~260MB total).
  const size_t mat = (size_t)kB * kM * kN;
  __bf16* qb       = (__bf16*)d_ws;
  __bf16* kb       = qb + mat;
  __bf16* vb       = kb + mat;
  __bf16* attn     = vb + mat;
  float*  ctx_part = (float*)(attn + mat);
  __bf16* ctxT     = (__bf16*)(ctx_part + (size_t)kSplit * kBH * kKD * kVD);
  float*  rmax     = (float*)(ctxT + (size_t)kBH * kKD * kVD);
  float*  rinv     = rmax + kRows;

  const dim3 blk(256);
  const dim3 gQKV(kN / 128, kM / 128, kB);
  qkv_proj_kernel<<<gQKV, blk, 0, stream>>>(x, wq, bq, qb);
  qkv_proj_kernel<<<gQKV, blk, 0, stream>>>(x, wk, bk, kb);
  qkv_proj_kernel<<<gQKV, blk, 0, stream>>>(x, wv, bv, vb);

  const int cols = kBH * kN;
  ln_kernel<<<cols / 256, blk, 0, stream>>>(qb, g_q, be_q, 1);
  ln_kernel<<<cols / 256, blk, 0, stream>>>(kb, g_k, be_k, 0);
  ln_kernel<<<cols / 256, blk, 0, stream>>>(vb, g_v, be_v, 0);

  k_rowstats_kernel<<<kRows, blk, 0, stream>>>(kb, rmax, rinv);
  ctx_gemm_kernel<<<dim3(kSplit, kBH), blk, 0, stream>>>(kb, vb, rmax, rinv, ctx_part);
  ctx_reduce_kernel<<<kBH, blk, 0, stream>>>(ctx_part, ctxT);
  attn_out_kernel<<<dim3(kN / 128, kBH), blk, 0, stream>>>(ctxT, qb, attn);
  out_proj_kernel<<<dim3(kN / 128, kC / 128, kB), blk, 0, stream>>>(attn, wo, bo,
                                                                    (float*)d_out);
}
